// CrossLinearNO_29927332119149
// MI455X (gfx1250) — matrix operand order repaired and checked by reading, not compiled, not run on hardware
//
#include <hip/hip_runtime.h>
#include <hip/hip_bf16.h>

typedef __attribute__((ext_vector_type(16))) _Float16 v16h;
typedef __attribute__((ext_vector_type(8)))  float    v8f;

union FragA { v16h v; _Float16 h[16]; };
union FragC { v8f  v; float    f[8];  };

constexpr int cB = 4, cN = 4096, cM = 4096, cDIM = 256;
constexpr int cH = 8, cDH = 64, cINNER = 512, cDK = 256, cBH = cB * cH;

// ---------------------------------------------------------------------------
// WMMA helpers (gfx1250 16x16x32 f16 -> f32)
// ---------------------------------------------------------------------------
__device__ __forceinline__ v8f wmma_f16(v16h a, v16h b, v8f c) {
  return __builtin_amdgcn_wmma_f32_16x16x32_f16(false, a, false, b, (short)0, c,
                                                false, false);
}

// A (or B) fragment from a row-major LDS tile: logical element (row0+r, k0+k).
__device__ __forceinline__ v16h ld_frag(const _Float16* lds, int row0, int k0,
                                        int stride) {
  const int lane = threadIdx.x & 31;
  const int hl = lane >> 4;
  const int r = lane & 15;
  FragA f;
  const _Float16* p = lds + (row0 + r) * stride + k0 + hl * 8;
#pragma unroll
  for (int i = 0; i < 8; ++i) f.h[i] = p[i];
#pragma unroll
  for (int i = 0; i < 8; ++i) f.h[8 + i] = p[16 + i];
  return f.v;
}

// Transposed-A fragment: logical A[row][k] stored in LDS as tile[k][row].
__device__ __forceinline__ v16h ld_frag_T(const _Float16* lds, int row0, int k0,
                                          int stride) {
  const int lane = threadIdx.x & 31;
  const int hl = lane >> 4;
  const int r = lane & 15;
  FragA f;
#pragma unroll
  for (int i = 0; i < 8; ++i)
    f.h[i] = lds[(k0 + hl * 8 + i) * stride + row0 + r];
#pragma unroll
  for (int i = 0; i < 8; ++i)
    f.h[8 + i] = lds[(k0 + hl * 8 + 16 + i) * stride + row0 + r];
  return f.v;
}

__device__ __forceinline__ void atomicMaxF(float* addr, float val) {
  if (val >= 0.f)
    atomicMax((int*)addr, __float_as_int(val));
  else
    atomicMin((unsigned int*)addr, __float_as_uint(val));
}

// ---------------------------------------------------------------------------
// Tile stager for the big GEMM. f16 activations go memory->LDS via the
// gfx1250 async-tensor path (global_load_async_to_lds_b128, ASYNCcnt);
// fp32 inputs take the load+convert path. Row stride 72 halves = 144 B
// (multiple of 16 B) keeps every async LDS target 16B-aligned.
// ---------------------------------------------------------------------------
template <typename TA>
__device__ __forceinline__ void stage_tiles(const TA* __restrict__ A,
                                            const float* __restrict__ W,
                                            _Float16* sAbuf, _Float16* sBbuf,
                                            int m0, int n0, int K, int k0,
                                            int tid) {
  if (sizeof(TA) == 2) {
    // 128 rows x 64 k f16 = 128 x 8 chunks of 16B; 256 lanes x 4 issues
    unsigned ldsbase = (unsigned)(size_t)sAbuf;
#pragma unroll
    for (int i = 0; i < 4; ++i) {
      int idx = tid + i * 256;
      int row = idx >> 3, ch = idx & 7;
      unsigned laddr = ldsbase + (unsigned)(row * 144 + ch * 16);
      unsigned long long ga = (unsigned long long)(size_t)(
          A + (size_t)(m0 + row) * K + k0 + ch * 8);
      asm volatile("global_load_async_to_lds_b128 %0, %1, off" ::"v"(laddr),
                   "v"(ga)
                   : "memory");
    }
  } else {
#pragma unroll
    for (int i = 0; i < 32; ++i) {
      int idx = tid + i * 256;
      int row = idx >> 6, kk = idx & 63;
      sAbuf[row * 72 + kk] = (_Float16)A[(size_t)(m0 + row) * K + k0 + kk];
    }
  }
  // W tile 64x64, fp32 -> f16
#pragma unroll
  for (int i = 0; i < 16; ++i) {
    int idx = tid + i * 256;
    int row = idx >> 6, kk = idx & 63;
    sBbuf[row * 72 + kk] = (_Float16)W[(size_t)(n0 + row) * K + k0 + kk];
  }
}

// ---------------------------------------------------------------------------
// Generic tiled WMMA GEMM: out[M,Nout] = act(A[M,K] @ W[Nout,K]^T + bias)
// Block tile 128x64, 8 waves, wave = 32x32 accumulator (2x2 fragments).
// Double-buffered 64-wide K stages: 8 WMMAs per barrier pair.
// ---------------------------------------------------------------------------
template <typename TA, typename TO, bool GELU>
__global__ __launch_bounds__(256) void gemm_bias_act(
    const TA* __restrict__ A, const float* __restrict__ W,
    const float* __restrict__ bias, TO* __restrict__ out, int Mrows, int K,
    int Nout) {
  __shared__ _Float16 sA[2][128 * 72];
  __shared__ _Float16 sB[2][64 * 72];
  const int tid = threadIdx.x;
  const int wave = tid >> 5;
  const int lane = tid & 31;
  const int hl = lane >> 4;
  const int r = lane & 15;
  const int m0 = blockIdx.x * 128;
  const int n0 = blockIdx.y * 64;
  const int wm = (wave >> 1) * 32;  // 0,32,64,96
  const int wn = (wave & 1) * 32;   // 0,32

  FragC acc[2][2];
#pragma unroll
  for (int i = 0; i < 2; ++i)
#pragma unroll
    for (int j = 0; j < 2; ++j)
#pragma unroll
      for (int e = 0; e < 8; ++e) acc[i][j].f[e] = 0.f;

  const int nst = K / 64;
  stage_tiles(A, W, sA[0], sB[0], m0, n0, K, 0, tid);
  for (int s = 0; s < nst; ++s) {
    if (s + 1 < nst)
      stage_tiles(A, W, sA[(s + 1) & 1], sB[(s + 1) & 1], m0, n0, K,
                  (s + 1) * 64, tid);
    if (sizeof(TA) == 2) {
      // wait for current stage's 4 async copies; let next stage's ride
      if (s + 1 < nst)
        asm volatile("s_wait_asynccnt 0x4" ::: "memory");
      else
        asm volatile("s_wait_asynccnt 0x0" ::: "memory");
    }
    __syncthreads();
    const _Float16* cA = sA[s & 1];
    const _Float16* cB = sB[s & 1];
#pragma unroll
    for (int kk = 0; kk < 2; ++kk) {
      v16h af[2], bf[2];
#pragma unroll
      for (int i = 0; i < 2; ++i) af[i] = ld_frag(cA, wm + i * 16, kk * 32, 72);
#pragma unroll
      for (int j = 0; j < 2; ++j) bf[j] = ld_frag(cB, wn + j * 16, kk * 32, 72);
#pragma unroll
      for (int i = 0; i < 2; ++i)
#pragma unroll
        for (int j = 0; j < 2; ++j)
          acc[i][j].v = wmma_f16(af[i], bf[j], acc[i][j].v);
    }
    __syncthreads();
  }

#pragma unroll
  for (int i = 0; i < 2; ++i)
#pragma unroll
    for (int j = 0; j < 2; ++j)
#pragma unroll
      for (int e = 0; e < 8; ++e) {
        int row = m0 + wm + i * 16 + hl * 8 + e;
        int col = n0 + wn + j * 16 + r;
        if (row < Mrows && col < Nout) {
          float v = acc[i][j].f[e] + bias[col];
          if (GELU) v = 0.5f * v * (1.f + erff(v * 0.70710678118654752f));
          out[(size_t)row * Nout + col] = (TO)v;
        }
      }
}

// ---------------------------------------------------------------------------
// init: colmax=-inf, colsum=0, kv=0
// ---------------------------------------------------------------------------
__global__ void init_stats(float* colmax, float* colsum, float* kvg) {
  int i = blockIdx.x * 256 + threadIdx.x;
  if (i < cBH * cDK) {
    colmax[i] = -3.0e38f;
    colsum[i] = 0.f;
  }
  if (i < cBH * cDK * cDH) kvg[i] = 0.f;
}

// ---------------------------------------------------------------------------
// Pass 1: column max of k-logits (y_mid @ Wk^T), per (b,h).
// Block: 128 M-rows x 256 cols; wave w owns 32 cols.
// ---------------------------------------------------------------------------
__global__ __launch_bounds__(256) void kstats_max(
    const _Float16* __restrict__ midy, const float* __restrict__ Wk,
    float* __restrict__ colmax) {
  __shared__ _Float16 sY[128 * 72];
  __shared__ _Float16 sWk[256 * 66];
  const int tid = threadIdx.x;
  const int wave = tid >> 5;
  const int lane = tid & 31;
  const int hl = lane >> 4;
  const int r = lane & 15;
  const int bh = blockIdx.y, b = bh >> 3, h = bh & 7;
  const int m0 = blockIdx.x * 128;

#pragma unroll
  for (int i = 0; i < 64; ++i) {
    int idx = tid + i * 256;
    int nn = idx >> 6, k = idx & 63;
    sWk[nn * 66 + k] = (_Float16)Wk[nn * 64 + k];
  }
#pragma unroll
  for (int i = 0; i < 32; ++i) {
    int idx = tid + i * 256;
    int m = idx >> 6, k = idx & 63;
    sY[m * 72 + k] = midy[(size_t)(b * cM + m0 + m) * cINNER + h * cDH + k];
  }
  __syncthreads();

  const int wcol = wave * 32;
  float cmax[2] = {-3.0e38f, -3.0e38f};
  for (int rc = 0; rc < 4; ++rc) {
    FragC acc[2][2];
#pragma unroll
    for (int i = 0; i < 2; ++i)
#pragma unroll
      for (int j = 0; j < 2; ++j)
#pragma unroll
        for (int e = 0; e < 8; ++e) acc[i][j].f[e] = 0.f;
#pragma unroll
    for (int ks = 0; ks < 2; ++ks) {
      v16h af[2], bf[2];
#pragma unroll
      for (int fm = 0; fm < 2; ++fm)
        af[fm] = ld_frag(sY, rc * 32 + fm * 16, ks * 32, 72);
#pragma unroll
      for (int fn = 0; fn < 2; ++fn)
        bf[fn] = ld_frag(sWk, wcol + fn * 16, ks * 32, 66);
#pragma unroll
      for (int fm = 0; fm < 2; ++fm)
#pragma unroll
        for (int fn = 0; fn < 2; ++fn)
          acc[fm][fn].v = wmma_f16(af[fm], bf[fn], acc[fm][fn].v);
    }
#pragma unroll
    for (int fm = 0; fm < 2; ++fm)
#pragma unroll
      for (int fn = 0; fn < 2; ++fn)
#pragma unroll
        for (int e = 0; e < 8; ++e)
          cmax[fn] = fmaxf(cmax[fn], acc[fm][fn].f[e]);
  }
  // lanes L and L^16 hold the same column (different rows): combine halves
#pragma unroll
  for (int fn = 0; fn < 2; ++fn) {
    float v = fmaxf(cmax[fn], __shfl_xor(cmax[fn], 16));
    if (hl == 0) atomicMaxF(&colmax[bh * cDK + wcol + fn * 16 + r], v);
  }
}

// ---------------------------------------------------------------------------
// Pass 2: kv += exp(klog - colmax)^T @ v ; colsum += column sums of exp tile.
// Block handles one (b,h) and a 32-row M chunk.
// ---------------------------------------------------------------------------
__global__ __launch_bounds__(256) void kv_accum(
    const _Float16* __restrict__ midy, const float* __restrict__ Wk,
    const float* __restrict__ Wv, const float* __restrict__ colmax,
    float* __restrict__ colsum, float* __restrict__ kvg) {
  __shared__ _Float16 sY[32 * 72];
  __shared__ _Float16 sW[256 * 66];   // Wk, later reused for Wv
  __shared__ _Float16 sAe[32 * 264];  // exp tile [m][d]
  __shared__ _Float16 sV[32 * 72];    // v tile   [m][c]
  const int tid = threadIdx.x;
  const int wave = tid >> 5;
  const int lane = tid & 31;
  const int hl = lane >> 4;
  const int r = lane & 15;
  const int bh = blockIdx.y, b = bh >> 3, h = bh & 7;
  const int m0 = blockIdx.x * 32;

#pragma unroll
  for (int i = 0; i < 64; ++i) {
    int idx = tid + i * 256;
    int nn = idx >> 6, k = idx & 63;
    sW[nn * 66 + k] = (_Float16)Wk[nn * 64 + k];
  }
#pragma unroll
  for (int i = 0; i < 8; ++i) {
    int idx = tid + i * 256;
    int m = idx >> 6, k = idx & 63;
    sY[m * 72 + k] = midy[(size_t)(b * cM + m0 + m) * cINNER + h * cDH + k];
  }
  __syncthreads();

  // klog [32 x 256], wave owns 32 cols -> exp tile into sAe
  {
    const int wcol = wave * 32;
    FragC acc[2][2];
#pragma unroll
    for (int i = 0; i < 2; ++i)
#pragma unroll
      for (int j = 0; j < 2; ++j)
#pragma unroll
        for (int e = 0; e < 8; ++e) acc[i][j].f[e] = 0.f;
#pragma unroll
    for (int ks = 0; ks < 2; ++ks) {
      v16h af[2], bf[2];
#pragma unroll
      for (int fm = 0; fm < 2; ++fm) af[fm] = ld_frag(sY, fm * 16, ks * 32, 72);
#pragma unroll
      for (int fn = 0; fn < 2; ++fn)
        bf[fn] = ld_frag(sW, wcol + fn * 16, ks * 32, 66);
#pragma unroll
      for (int fm = 0; fm < 2; ++fm)
#pragma unroll
        for (int fn = 0; fn < 2; ++fn)
          acc[fm][fn].v = wmma_f16(af[fm], bf[fn], acc[fm][fn].v);
    }
#pragma unroll
    for (int fm = 0; fm < 2; ++fm)
#pragma unroll
      for (int fn = 0; fn < 2; ++fn)
#pragma unroll
        for (int e = 0; e < 8; ++e) {
          int row = fm * 16 + hl * 8 + e;
          int col = wcol + fn * 16 + r;
          float ev = expf(acc[fm][fn].f[e] - colmax[bh * cDK + col]);
          sAe[row * 264 + col] = (_Float16)ev;
        }
  }
  __syncthreads();

  // column-sum partials + overwrite sW with Wv
  {
    int col = tid;  // 256 threads, 256 columns
    float s = 0.f;
#pragma unroll
    for (int m = 0; m < 32; ++m) s += (float)sAe[m * 264 + col];
    atomicAdd(&colsum[bh * cDK + col], s);
  }
#pragma unroll
  for (int i = 0; i < 16; ++i) {
    int idx = tid + i * 256;
    int nn = idx >> 6, k = idx & 63;
    sW[nn * 66 + k] = (_Float16)Wv[nn * 64 + k];
  }
  __syncthreads();

  // v = y_mid @ Wv^T  [32 x 64] -> sV; 8 waves, 1 fragment each
  {
    const int r0 = (wave >> 2) * 16;
    const int c0 = (wave & 3) * 16;
    FragC o;
#pragma unroll
    for (int e = 0; e < 8; ++e) o.f[e] = 0.f;
#pragma unroll
    for (int ks = 0; ks < 2; ++ks) {
      v16h af = ld_frag(sY, r0, ks * 32, 72);
      v16h bf = ld_frag(sW, c0, ks * 32, 66);
      o.v = wmma_f16(af, bf, o.v);
    }
#pragma unroll
    for (int e = 0; e < 8; ++e)
      sV[(r0 + hl * 8 + e) * 72 + c0 + r] = (_Float16)o.f[e];
  }
  __syncthreads();

  // kv partial = sAe^T [256x32] @ sV [32x64]; wave owns 32 d-rows x 64 cols
  {
    const int d0 = wave * 32;
    v16h af[2];
#pragma unroll
    for (int fd = 0; fd < 2; ++fd)
      af[fd] = ld_frag_T(sAe, d0 + fd * 16, 0, 264);
#pragma unroll
    for (int fc = 0; fc < 4; ++fc) {
      v16h bf = ld_frag_T(sV, fc * 16, 0, 72);
#pragma unroll
      for (int fd = 0; fd < 2; ++fd) {
        FragC o;
#pragma unroll
        for (int e = 0; e < 8; ++e) o.f[e] = 0.f;
        o.v = wmma_f16(af[fd], bf, o.v);
#pragma unroll
        for (int e = 0; e < 8; ++e) {
          int d = d0 + fd * 16 + hl * 8 + e;
          int c = fc * 16 + r;
          atomicAdd(&kvg[(size_t)bh * cDK * cDH + d * cDH + c], o.f[e]);
        }
      }
    }
  }
}

// ---------------------------------------------------------------------------
// q-logits -> row softmax -> @ (kv / colsum) -> qkv written as [B,N,INNER]
// Block handles one (b,h) and a 32-row N chunk.
// ---------------------------------------------------------------------------
__global__ __launch_bounds__(256) void qkv_kernel(
    const _Float16* __restrict__ midx, const float* __restrict__ Wq,
    const float* __restrict__ kvg, const float* __restrict__ colsum,
    _Float16* __restrict__ qkvb) {
  __shared__ _Float16 sX[32 * 72];
  __shared__ _Float16 sWB[256 * 66];  // Wq, later kv^T as [c][d] stride 264
  __shared__ _Float16 sQ[32 * 264];
  __shared__ float red[32 * 8];
  __shared__ float rowstat[32];
  const int tid = threadIdx.x;
  const int wave = tid >> 5;
  const int lane = tid & 31;
  const int hl = lane >> 4;
  const int r = lane & 15;
  const int bh = blockIdx.y, b = bh >> 3, h = bh & 7;
  const int n0 = blockIdx.x * 32;

#pragma unroll
  for (int i = 0; i < 8; ++i) {
    int idx = tid + i * 256;
    int row = idx >> 6, k = idx & 63;
    sX[row * 72 + k] = midx[(size_t)(b * cN + n0 + row) * cINNER + h * cDH + k];
  }
#pragma unroll
  for (int i = 0; i < 64; ++i) {
    int idx = tid + i * 256;
    int nn = idx >> 6, k = idx & 63;
    sWB[nn * 66 + k] = (_Float16)Wq[nn * 64 + k];
  }
  __syncthreads();

  const int wcol = wave * 32;
  FragC acc[2][2];
#pragma unroll
  for (int i = 0; i < 2; ++i)
#pragma unroll
    for (int j = 0; j < 2; ++j)
#pragma unroll
      for (int e = 0; e < 8; ++e) acc[i][j].f[e] = 0.f;
#pragma unroll
  for (int ks = 0; ks < 2; ++ks) {
    v16h af[2], bf[2];
#pragma unroll
    for (int fm = 0; fm < 2; ++fm) af[fm] = ld_frag(sX, fm * 16, ks * 32, 72);
#pragma unroll
    for (int fn = 0; fn < 2; ++fn)
      bf[fn] = ld_frag(sWB, wcol + fn * 16, ks * 32, 66);
#pragma unroll
    for (int fm = 0; fm < 2; ++fm)
#pragma unroll
      for (int fn = 0; fn < 2; ++fn)
        acc[fm][fn].v = wmma_f16(af[fm], bf[fn], acc[fm][fn].v);
  }

  // row max (wave partial -> LDS -> combine)
#pragma unroll
  for (int fm = 0; fm < 2; ++fm)
#pragma unroll
    for (int e = 0; e < 8; ++e) {
      int row = fm * 16 + hl * 8 + e;
      float v = fmaxf(acc[fm][0].f[e], acc[fm][1].f[e]);
#pragma unroll
      for (int mk = 1; mk < 16; mk <<= 1) v = fmaxf(v, __shfl_xor(v, mk));
      if (r == 0) red[row * 8 + wave] = v;
    }
  __syncthreads();
  if (tid < 32) {
    float v = red[tid * 8];
#pragma unroll
    for (int w2 = 1; w2 < 8; ++w2) v = fmaxf(v, red[tid * 8 + w2]);
    rowstat[tid] = v;
  }
  __syncthreads();

  // exp + row sum
#pragma unroll
  for (int fm = 0; fm < 2; ++fm)
#pragma unroll
    for (int e = 0; e < 8; ++e) {
      int row = fm * 16 + hl * 8 + e;
      float rm = rowstat[row];
      float s = 0.f;
#pragma unroll
      for (int fn = 0; fn < 2; ++fn) {
        float ev = expf(acc[fm][fn].f[e] - rm);
        acc[fm][fn].f[e] = ev;
        s += ev;
      }
#pragma unroll
      for (int mk = 1; mk < 16; mk <<= 1) s += __shfl_xor(s, mk);
      if (r == 0) red[row * 8 + wave] = s;
    }
  __syncthreads();
  if (tid < 32) {
    float v = 0.f;
#pragma unroll
    for (int w2 = 0; w2 < 8; ++w2) v += red[tid * 8 + w2];
    rowstat[tid] = v;
  }
  __syncthreads();

  // normalize -> sQ; stage kv^T / colsum into sWB
#pragma unroll
  for (int fm = 0; fm < 2; ++fm)
#pragma unroll
    for (int fn = 0; fn < 2; ++fn)
#pragma unroll
      for (int e = 0; e < 8; ++e) {
        int row = fm * 16 + hl * 8 + e;
        int col = wcol + fn * 16 + r;
        sQ[row * 264 + col] = (_Float16)(acc[fm][fn].f[e] / rowstat[row]);
      }
#pragma unroll
  for (int i = 0; i < 64; ++i) {
    int idx = tid + i * 256;
    int c = idx >> 8, d = idx & 255;
    float kvv =
        kvg[(size_t)bh * cDK * cDH + d * cDH + c] / colsum[bh * cDK + d];
    sWB[c * 264 + d] = (_Float16)kvv;
  }
  __syncthreads();

  // qkv = sQ [32x256] @ kvhat [256x64]; 8 waves, one 16x16 tile each
  {
    const int r0 = (wave >> 2) * 16;
    const int c0 = (wave & 3) * 16;
    FragC o;
#pragma unroll
    for (int e = 0; e < 8; ++e) o.f[e] = 0.f;
#pragma unroll
    for (int ks = 0; ks < 8; ++ks) {
      v16h af = ld_frag(sQ, r0, ks * 32, 264);
      v16h bf = ld_frag(sWB, c0, ks * 32, 264);
      o.v = wmma_f16(af, bf, o.v);
    }
#pragma unroll
    for (int e = 0; e < 8; ++e) {
      int n = n0 + r0 + hl * 8 + e;
      int c = c0 + r;
      qkvb[(size_t)(b * cN + n) * cINNER + h * cDH + c] = (_Float16)o.f[e];
    }
  }
}

// ---------------------------------------------------------------------------
extern "C" void kernel_launch(void* const* d_in, const int* in_sizes, int n_in,
                              void* d_out, int out_size, void* d_ws,
                              size_t ws_size, hipStream_t stream) {
  (void)in_sizes; (void)n_in; (void)out_size; (void)ws_size;
  const float* x   = (const float*)d_in[0];
  const float* y   = (const float*)d_in[1];
  const float* Wx  = (const float*)d_in[2];
  const float* bx  = (const float*)d_in[3];
  const float* Wy  = (const float*)d_in[4];
  const float* by  = (const float*)d_in[5];
  const float* Wq  = (const float*)d_in[6];
  const float* Wk  = (const float*)d_in[7];
  const float* Wv  = (const float*)d_in[8];
  const float* Wo1 = (const float*)d_in[9];
  const float* bo1 = (const float*)d_in[10];
  const float* Wo2 = (const float*)d_in[11];
  const float* bo2 = (const float*)d_in[12];
  float* out = (float*)d_out;

  _Float16* midx = (_Float16*)d_ws;
  _Float16* midy = midx + (size_t)cB * cN * cINNER;
  _Float16* qkvb = midy + (size_t)cB * cM * cINNER;
  _Float16* h1   = qkvb + (size_t)cB * cN * cINNER;
  float* colmax  = (float*)(h1 + (size_t)cB * cN * cDIM);
  float* colsum  = colmax + cBH * cDK;
  float* kvg     = colsum + cBH * cDK;

  dim3 blk(256);
  // in-projections (fp32 in, f16 mid)
  gemm_bias_act<float, _Float16, false>
      <<<dim3((cB * cN) / 128, cINNER / 64), blk, 0, stream>>>(
          x, Wx, bx, midx, cB * cN, cDIM, cINNER);
  gemm_bias_act<float, _Float16, false>
      <<<dim3((cB * cM) / 128, cINNER / 64), blk, 0, stream>>>(
          y, Wy, by, midy, cB * cM, cDIM, cINNER);
  // k/v -> kv with token softmax
  init_stats<<<dim3((cBH * cDK * cDH + 255) / 256), blk, 0, stream>>>(
      colmax, colsum, kvg);
  kstats_max<<<dim3(cM / 128, cBH), blk, 0, stream>>>(midy, Wk, colmax);
  kv_accum<<<dim3(cM / 32, cBH), blk, 0, stream>>>(midy, Wk, Wv, colmax,
                                                   colsum, kvg);
  // q softmax @ kv
  qkv_kernel<<<dim3(cN / 32, cBH), blk, 0, stream>>>(midx, Wq, kvg, colsum,
                                                     qkvb);
  // out projections (f16 activations staged via async-to-LDS)
  gemm_bias_act<_Float16, _Float16, true>
      <<<dim3((cB * cN) / 128, cDIM / 64), blk, 0, stream>>>(
          qkvb, Wo1, bo1, h1, cB * cN, cINNER, cDIM);
  gemm_bias_act<_Float16, float, false>
      <<<dim3((cB * cN) / 128, cDIM / 64), blk, 0, stream>>>(
          h1, Wo2, bo2, out, cB * cN, cDIM, cDIM);
}
